// Seq2SeqLSTM_25486335934551
// MI455X (gfx1250) — compile-verified, matrix operand order checked by reading, not run on hardware
//
#include <hip/hip_runtime.h>

// ---------------------------------------------------------------------------
// Seq2Seq LSTM (V=32000, E=512, H=1024, B=32, S=T=64) for gfx1250 / MI455X.
// bf16 weights/activations, f32 accumulation via v_wmma_f32_16x16x32_bf16.
// Block-shared A operands are staged into LDS with the Tensor Data Mover
// (tensor_load_to_lds + s_wait_tensorcnt), then read via ds_load_b128.
// ---------------------------------------------------------------------------

typedef __bf16 bf16;
typedef __attribute__((ext_vector_type(8)))  __bf16 v8bf;
typedef __attribute__((ext_vector_type(16))) __bf16 v16bf;
typedef __attribute__((ext_vector_type(8)))  float  v8f;
typedef __attribute__((ext_vector_type(4)))  unsigned int u32x4;
typedef __attribute__((ext_vector_type(4)))  int i32x4;
typedef __attribute__((ext_vector_type(8)))  int i32x8;

#define LAUNCH(k, g, b) k<<<(g), (b), 0, stream>>>

static __device__ __forceinline__ bf16 f2bf(float f) {
  unsigned int u = __float_as_uint(f);
  unsigned int r = (u + 0x7FFFu + ((u >> 16) & 1u)) >> 16;
  unsigned short s = (unsigned short)r;
  bf16 b;
  __builtin_memcpy(&b, &s, 2);
  return b;
}

static __device__ __forceinline__ float sigmoidf_(float x) {
  return 1.0f / (1.0f + __expf(-x));
}

// ---------------------------------------------------------------------------
// TDM: DMA a 2-D tile (nrows x row_u32 dwords, densely packed) from global
// memory into LDS. Descriptor layout per CDNA5 ISA section 8.3/8.4:
//   group0: count=1 | lds_addr | global_addr[56:0] | type=2
//   group1: data_size=4B, tensor_dim0/1, tile_dim0/1, tensor_dim0_stride
//   groups 2/3: unused (2-D tensor)
// Issue from one wave; that wave s_wait_tensorcnt(0)s before the block
// barrier publishes the data.
// ---------------------------------------------------------------------------
static __device__ __forceinline__ void tdm_load_2d(unsigned lds_off,
                                                   const void* gptr,
                                                   unsigned row_u32,
                                                   unsigned nrows) {
  unsigned long long ga = (unsigned long long)(uintptr_t)gptr;
  u32x4 g0;
  g0[0] = 1u;                                    // count=1 (user descriptor)
  g0[1] = lds_off;                               // LDS byte address
  g0[2] = (unsigned)(ga & 0xFFFFFFFFu);          // global_addr[31:0]
  g0[3] = (unsigned)((ga >> 32) & 0x01FFFFFFu) | 0x80000000u;  // [56:32]|type=2
  i32x8 g1;
  g1[0] = (int)(2u << 16);                       // data_size = 4 bytes
  g1[1] = (int)(row_u32 << 16);                  // tensor_dim0 (low 16)
  g1[2] = (int)((row_u32 >> 16) | (nrows << 16));// dim0 hi | tensor_dim1 lo
  g1[3] = (int)(row_u32 << 16);                  // tile_dim0
  g1[4] = (int)(nrows & 0xFFFFu);                // tile_dim1
  g1[5] = (int)row_u32;                          // tensor_dim0_stride (low 32)
  g1[6] = 0;
  g1[7] = 0;
  i32x4 z4 = {};
#if defined(__clang_major__) && (__clang_major__ >= 23)
  i32x8 z8 = {};
  __builtin_amdgcn_tensor_load_to_lds(g0, g1, z4, z4, z8, 0);
#else
  __builtin_amdgcn_tensor_load_to_lds(g0, g1, z4, z4, 0);
#endif
}

// Load one 16x32 bf16 tile (A operand, or B operand stored as 16 rows x K,
// row-major K-contiguous). Wave32 layout per CDNA5 ISA:
// lanes 0-15 : row = lane,     K = [k0..k0+7] ++ [k0+16..k0+23]
// lanes 16-31: row = lane-16,  K = [k0+8..k0+15] ++ [k0+24..k0+31]
static __device__ __forceinline__ v16bf load_tile(const bf16* __restrict__ base,
                                                  int ld, int row0, int k0,
                                                  int lane) {
  int m = lane & 15;
  int ks = (lane >> 4) << 3;  // 0 or 8
  const bf16* p = base + (size_t)(row0 + m) * ld + (k0 + ks);
  v8bf lo = *reinterpret_cast<const v8bf*>(p);
  v8bf hi = *reinterpret_cast<const v8bf*>(p + 16);
  return __builtin_shufflevector(lo, hi, 0, 1, 2, 3, 4, 5, 6, 7, 8, 9, 10, 11,
                                 12, 13, 14, 15);
}

static __device__ __forceinline__ v8f wmma_bf16(v16bf a, v16bf b, v8f c) {
  return __builtin_amdgcn_wmma_f32_16x16x32_bf16(false, a, false, b, (short)0,
                                                 c, false, false);
}

// ---------------------------------------------------------------------------
// Small utility kernels
// ---------------------------------------------------------------------------
__global__ void k_f32_to_bf16(const float* __restrict__ s, bf16* __restrict__ d,
                              int n) {
  int i = blockIdx.x * blockDim.x + threadIdx.x;
  if (i < n) d[i] = f2bf(s[i]);
}

// Gather embeddings into bf16 X matrix, rows ordered r = step*32 + batch.
// mode 0: encoder ids = input_seq[b, t]
// mode 1: decoder ids = (t==0) ? SOS(=1) : target_seq[b, t-1]
__global__ void k_embed(const int* __restrict__ seq,
                        const float* __restrict__ emb, bf16* __restrict__ dst,
                        int mode) {
  int idx = blockIdx.x * blockDim.x + threadIdx.x;  // 2048 * 512
  int row = idx >> 9;
  int col = idx & 511;
  int t = row >> 5;
  int b = row & 31;
  int id;
  if (mode == 0)
    id = seq[b * 64 + t];
  else
    id = (t == 0) ? 1 : seq[b * 64 + t - 1];
  dst[idx] = f2bf(emb[(size_t)id * 512 + col]);
}

__global__ void k_zero_state(bf16* h0a, bf16* h0b, bf16* h1a, bf16* h1b,
                             float* c0, float* c1) {
  int i = blockIdx.x * blockDim.x + threadIdx.x;  // 32*1024
  bf16 z = f2bf(0.0f);
  h0a[i] = z;
  h0b[i] = z;
  h1a[i] = z;
  h1b[i] = z;
  c0[i] = 0.0f;
  c1[i] = 0.0f;
}

// ---------------------------------------------------------------------------
// gx = X @ Wih0^T + (bih0 + bhh0)   for all 64 steps at once.
// X: [2048, 512] bf16,  W: [4096, 512] bf16,  gx: [2048, 4096] f32.
// A tile (16 x 512) staged via TDM into LDS, shared by the 8 waves.
// grid = (32 n-blocks, 128 m-tiles), block = 256.
// ---------------------------------------------------------------------------
__global__ void k_gates_x(const bf16* __restrict__ X,
                          const bf16* __restrict__ W,
                          const float* __restrict__ b0,
                          const float* __restrict__ b1,
                          float* __restrict__ gx) {
  __shared__ __align__(16) bf16 shA[16 * 512];  // 16 KB
  int lane = threadIdx.x & 31;
  int wv = threadIdx.x >> 5;
  int n0 = (blockIdx.x * 8 + wv) * 16;
  int m0 = blockIdx.y * 16;
  if (threadIdx.x < 32) {
    tdm_load_2d((unsigned)(uintptr_t)&shA[0], X + (size_t)m0 * 512, 256, 16);
    __builtin_amdgcn_s_wait_tensorcnt(0);
  }
  __syncthreads();
  v8f acc = {};
  for (int k0 = 0; k0 < 512; k0 += 32) {
    v16bf a = load_tile(shA, 512, 0, k0, lane);
    v16bf b = load_tile(W, 512, n0, k0, lane);
    acc = wmma_bf16(a, b, acc);
  }
  int n = n0 + (lane & 15);
  int mb = m0 + ((lane >> 4) << 3);
  float bias = b0[n] + b1[n];
#pragma unroll
  for (int r = 0; r < 8; ++r)
    gx[(size_t)(mb + r) * 4096 + n] = acc[r] + bias;
}

// ---------------------------------------------------------------------------
// LSTM layer-0 step: gates = gx(precomputed, incl. biases) + h_in @ Whh^T.
// h_in (32x1024 bf16 = 64 KB) is TDM-staged into LDS once per block and
// shared by all 8 waves. 128 wave-tasks = (2 batch halves) x (64 j-tiles);
// each wave keeps 4 gate accumulators sharing the A operand and performs the
// i/f/g/o cell update entirely in registers. c in place; h ping-pongs.
// ---------------------------------------------------------------------------
__global__ void k_lstm_l0(const bf16* __restrict__ h_in,
                          const bf16* __restrict__ Whh,
                          const float* __restrict__ gx, float* __restrict__ c,
                          bf16* __restrict__ h_out) {
  __shared__ __align__(16) bf16 shA[32 * 1024];  // 64 KB
  int lane = threadIdx.x & 31;
  int task = blockIdx.x * 8 + (threadIdx.x >> 5);  // 0..127
  int m_tile = task & 1;
  int j_tile = task >> 1;
  int rowA = m_tile * 16;
  if (threadIdx.x < 32) {
    tdm_load_2d((unsigned)(uintptr_t)&shA[0], h_in, 512, 32);
    __builtin_amdgcn_s_wait_tensorcnt(0);
  }
  __syncthreads();
  v8f acc[4] = {};
  for (int k0 = 0; k0 < 1024; k0 += 32) {
    v16bf a = load_tile(shA, 1024, rowA, k0, lane);
#pragma unroll
    for (int g = 0; g < 4; ++g) {
      v16bf b = load_tile(Whh, 1024, g * 1024 + j_tile * 16, k0, lane);
      acc[g] = wmma_bf16(a, b, acc[g]);
    }
  }
  int n = j_tile * 16 + (lane & 15);
  int mbase = rowA + ((lane >> 4) << 3);
#pragma unroll
  for (int r = 0; r < 8; ++r) {
    int m = mbase + r;
    size_t gi = (size_t)m * 4096;
    float ig = acc[0][r] + gx[gi + n];
    float fg = acc[1][r] + gx[gi + 1024 + n];
    float gg = acc[2][r] + gx[gi + 2048 + n];
    float og = acc[3][r] + gx[gi + 3072 + n];
    size_t ci = (size_t)m * 1024 + n;
    float cn = sigmoidf_(fg) * c[ci] + sigmoidf_(ig) * tanhf(gg);
    c[ci] = cn;
    h_out[ci] = f2bf(sigmoidf_(og) * tanhf(cn));
  }
}

// ---------------------------------------------------------------------------
// LSTM layer-1 step: gates = h0 @ Wih1^T + h1_in @ Whh1^T + bih1 + bhh1.
// One 64 KB LDS buffer reused for both A matrices (h0 phase, then h1 phase)
// with barriers separating the phases. Optionally stores the new top-layer
// hidden into hs (decoder outputs).
// ---------------------------------------------------------------------------
__global__ void k_lstm_l1(const bf16* __restrict__ h0,
                          const bf16* __restrict__ h1_in,
                          const bf16* __restrict__ Wih,
                          const bf16* __restrict__ Whh,
                          const float* __restrict__ b0,
                          const float* __restrict__ b1, float* __restrict__ c,
                          bf16* __restrict__ h1_out, bf16* __restrict__ hs) {
  __shared__ __align__(16) bf16 shA[32 * 1024];  // 64 KB, reused twice
  int lane = threadIdx.x & 31;
  int task = blockIdx.x * 8 + (threadIdx.x >> 5);
  int m_tile = task & 1;
  int j_tile = task >> 1;
  int rowA = m_tile * 16;
  v8f acc[4] = {};
  // ---- phase 1: input-side GEMM with A = h0 ----
  if (threadIdx.x < 32) {
    tdm_load_2d((unsigned)(uintptr_t)&shA[0], h0, 512, 32);
    __builtin_amdgcn_s_wait_tensorcnt(0);
  }
  __syncthreads();
  for (int k0 = 0; k0 < 1024; k0 += 32) {
    v16bf a = load_tile(shA, 1024, rowA, k0, lane);
#pragma unroll
    for (int g = 0; g < 4; ++g) {
      v16bf b = load_tile(Wih, 1024, g * 1024 + j_tile * 16, k0, lane);
      acc[g] = wmma_bf16(a, b, acc[g]);
    }
  }
  // ---- phase 2: recurrent GEMM with A = h1_in ----
  __syncthreads();  // everyone done reading shA before overwrite
  if (threadIdx.x < 32) {
    tdm_load_2d((unsigned)(uintptr_t)&shA[0], h1_in, 512, 32);
    __builtin_amdgcn_s_wait_tensorcnt(0);
  }
  __syncthreads();
  for (int k0 = 0; k0 < 1024; k0 += 32) {
    v16bf a = load_tile(shA, 1024, rowA, k0, lane);
#pragma unroll
    for (int g = 0; g < 4; ++g) {
      v16bf b = load_tile(Whh, 1024, g * 1024 + j_tile * 16, k0, lane);
      acc[g] = wmma_bf16(a, b, acc[g]);
    }
  }
  int n = j_tile * 16 + (lane & 15);
  int mbase = rowA + ((lane >> 4) << 3);
#pragma unroll
  for (int r = 0; r < 8; ++r) {
    int m = mbase + r;
    float ig = acc[0][r] + b0[n] + b1[n];
    float fg = acc[1][r] + b0[1024 + n] + b1[1024 + n];
    float gg = acc[2][r] + b0[2048 + n] + b1[2048 + n];
    float og = acc[3][r] + b0[3072 + n] + b1[3072 + n];
    size_t ci = (size_t)m * 1024 + n;
    float cn = sigmoidf_(fg) * c[ci] + sigmoidf_(ig) * tanhf(gg);
    c[ci] = cn;
    bf16 hb = f2bf(sigmoidf_(og) * tanhf(cn));
    h1_out[ci] = hb;
    if (hs) hs[ci] = hb;
  }
}

// ---------------------------------------------------------------------------
// Final FC: logits[b,t,v] = hs[t*32+b, :] . fc_W[v, :] + fc_b[v].
// hs: [2048,1024] bf16, fc_W: [32000,1024] bf16, out: [32,64,32000] f32.
// A tile (16x1024, 32 KB) TDM-staged into LDS, shared by 8 waves.
// grid = (250 n-blocks x 8 waves = 2000 n-tiles, 128 m-tiles).
// ---------------------------------------------------------------------------
__global__ void k_fc(const bf16* __restrict__ hs, const bf16* __restrict__ Wf,
                     const float* __restrict__ bias, float* __restrict__ out) {
  __shared__ __align__(16) bf16 shA[16 * 1024];  // 32 KB
  int lane = threadIdx.x & 31;
  int wv = threadIdx.x >> 5;
  int n0 = (blockIdx.x * 8 + wv) * 16;
  int m0 = blockIdx.y * 16;
  if (threadIdx.x < 32) {
    tdm_load_2d((unsigned)(uintptr_t)&shA[0], hs + (size_t)m0 * 1024, 512, 16);
    __builtin_amdgcn_s_wait_tensorcnt(0);
  }
  __syncthreads();
  v8f acc = {};
  for (int k0 = 0; k0 < 1024; k0 += 32) {
    __builtin_prefetch(Wf + (size_t)(n0 + (lane & 15)) * 1024 + k0 + 128, 0, 1);
    v16bf a = load_tile(shA, 1024, 0, k0, lane);
    v16bf b = load_tile(Wf, 1024, n0, k0, lane);
    acc = wmma_bf16(a, b, acc);
  }
  int v = n0 + (lane & 15);
  int mbase = m0 + ((lane >> 4) << 3);
  float bb = bias[v];
#pragma unroll
  for (int r = 0; r < 8; ++r) {
    int m = mbase + r;  // m = t*32 + b
    int t = m >> 5;
    int bidx = m & 31;
    out[((size_t)bidx * 64 + t) * 32000 + v] = acc[r] + bb;
  }
}

// ---------------------------------------------------------------------------
extern "C" void kernel_launch(void* const* d_in, const int* in_sizes, int n_in,
                              void* d_out, int out_size, void* d_ws,
                              size_t ws_size, hipStream_t stream) {
  (void)in_sizes; (void)n_in; (void)out_size; (void)ws_size;
  const int*   input_seq  = (const int*)d_in[0];
  const int*   target_seq = (const int*)d_in[1];
  const float* emb        = (const float*)d_in[2];
  const float* fc_W       = (const float*)d_in[3];
  const float* fc_b       = (const float*)d_in[4];
  const float* eWih0 = (const float*)d_in[5];
  const float* eWhh0 = (const float*)d_in[6];
  const float* eBih0 = (const float*)d_in[7];
  const float* eBhh0 = (const float*)d_in[8];
  const float* eWih1 = (const float*)d_in[9];
  const float* eWhh1 = (const float*)d_in[10];
  const float* eBih1 = (const float*)d_in[11];
  const float* eBhh1 = (const float*)d_in[12];
  const float* dWih0 = (const float*)d_in[13];
  const float* dWhh0 = (const float*)d_in[14];
  const float* dBih0 = (const float*)d_in[15];
  const float* dBhh0 = (const float*)d_in[16];
  const float* dWih1 = (const float*)d_in[17];
  const float* dWhh1 = (const float*)d_in[18];
  const float* dBih1 = (const float*)d_in[19];
  const float* dBhh1 = (const float*)d_in[20];

  char* w = (char*)d_ws;
  size_t off = 0;
  auto alloc = [&](size_t elems, size_t esz) -> void* {
    void* p = w + off;
    off += (elems * esz + 255) & ~(size_t)255;
    return p;
  };
  // bf16 weight copies
  bf16* wEih0 = (bf16*)alloc(4096 * 512, 2);
  bf16* wEhh0 = (bf16*)alloc(4096 * 1024, 2);
  bf16* wEih1 = (bf16*)alloc(4096 * 1024, 2);
  bf16* wEhh1 = (bf16*)alloc(4096 * 1024, 2);
  bf16* wDih0 = (bf16*)alloc(4096 * 512, 2);
  bf16* wDhh0 = (bf16*)alloc(4096 * 1024, 2);
  bf16* wDih1 = (bf16*)alloc(4096 * 1024, 2);
  bf16* wDhh1 = (bf16*)alloc(4096 * 1024, 2);
  bf16* wFc   = (bf16*)alloc((size_t)32000 * 1024, 2);
  // activations / state
  bf16* xinE = (bf16*)alloc(2048 * 512, 2);
  bf16* xinD = (bf16*)alloc(2048 * 512, 2);
  bf16* h0a  = (bf16*)alloc(32 * 1024, 2);
  bf16* h0b  = (bf16*)alloc(32 * 1024, 2);
  bf16* h1a  = (bf16*)alloc(32 * 1024, 2);
  bf16* h1b  = (bf16*)alloc(32 * 1024, 2);
  bf16* hsD  = (bf16*)alloc((size_t)2048 * 1024, 2);
  float* gxE = (float*)alloc((size_t)2048 * 4096, 4);
  float* gxD = (float*)alloc((size_t)2048 * 4096, 4);
  float* c0  = (float*)alloc(32 * 1024, 4);
  float* c1  = (float*)alloc(32 * 1024, 4);

  // 1) Weight conversion to bf16
  struct { const float* s; bf16* d; int n; } cv[9] = {
      {eWih0, wEih0, 4096 * 512},  {eWhh0, wEhh0, 4096 * 1024},
      {eWih1, wEih1, 4096 * 1024}, {eWhh1, wEhh1, 4096 * 1024},
      {dWih0, wDih0, 4096 * 512},  {dWhh0, wDhh0, 4096 * 1024},
      {dWih1, wDih1, 4096 * 1024}, {dWhh1, wDhh1, 4096 * 1024},
      {fc_W, wFc, 32000 * 1024}};
  for (int i = 0; i < 9; ++i)
    LAUNCH(k_f32_to_bf16, dim3((cv[i].n + 255) / 256), dim3(256))
    (cv[i].s, cv[i].d, cv[i].n);

  // 2) Embedding gather (enc + dec inputs)
  LAUNCH(k_embed, dim3(4096), dim3(256))(input_seq, emb, xinE, 0);
  LAUNCH(k_embed, dim3(4096), dim3(256))(target_seq, emb, xinD, 1);

  // 3) Precompute layer-0 input gates for all steps (bulk WMMA GEMM)
  LAUNCH(k_gates_x, dim3(32, 128), dim3(256))(xinE, wEih0, eBih0, eBhh0, gxE);
  LAUNCH(k_gates_x, dim3(32, 128), dim3(256))(xinD, wDih0, dBih0, dBhh0, gxD);

  // 4) Zero initial state
  LAUNCH(k_zero_state, dim3(128), dim3(256))(h0a, h0b, h1a, h1b, c0, c1);

  // 5) Encoder recurrence (h ping-pong, c in place)
  bf16 *h0_in = h0a, *h0_out = h0b, *h1_in = h1a, *h1_out = h1b;
  for (int s = 0; s < 64; ++s) {
    LAUNCH(k_lstm_l0, dim3(16), dim3(256))
    (h0_in, wEhh0, gxE + (size_t)s * 32 * 4096, c0, h0_out);
    LAUNCH(k_lstm_l1, dim3(16), dim3(256))
    (h0_out, h1_in, wEih1, wEhh1, eBih1, eBhh1, c1, h1_out, (bf16*)nullptr);
    bf16* t0 = h0_in; h0_in = h0_out; h0_out = t0;
    bf16* t1 = h1_in; h1_in = h1_out; h1_out = t1;
  }

  // 6) Decoder recurrence (state carries over from encoder), save top hiddens
  for (int t = 0; t < 64; ++t) {
    LAUNCH(k_lstm_l0, dim3(16), dim3(256))
    (h0_in, wDhh0, gxD + (size_t)t * 32 * 4096, c0, h0_out);
    LAUNCH(k_lstm_l1, dim3(16), dim3(256))
    (h0_out, h1_in, wDih1, wDhh1, dBih1, dBhh1, c1, h1_out,
     hsD + (size_t)t * 32 * 1024);
    bf16* t0 = h0_in; h0_in = h0_out; h0_out = t0;
    bf16* t1 = h1_in; h1_in = h1_out; h1_out = t1;
  }

  // 7) Vocab projection (bulk WMMA GEMM with fused bias + [B,T,V] permute)
  LAUNCH(k_fc, dim3(250, 128), dim3(256))(hsD, wFc, fc_b, (float*)d_out);
}
